// SpMiddleFHD_3178275799072
// MI455X (gfx1250) — matrix operand order repaired and checked by reading, not compiled, not run on hardware
//
#include <hip/hip_runtime.h>

// ---------------------------------------------------------------------------
// Types
// ---------------------------------------------------------------------------
typedef __bf16 bf16_t;
typedef __attribute__((ext_vector_type(16))) __bf16          v16bf;
typedef __attribute__((ext_vector_type(8)))  float           v8f;
typedef __attribute__((ext_vector_type(8)))  unsigned short  v8us;
typedef __attribute__((ext_vector_type(16))) unsigned short  v16us;

static __device__ __forceinline__ unsigned short f2bf(float f) {
    unsigned int u = __float_as_uint(f);
    unsigned int r = u + 0x7FFFu + ((u >> 16) & 1u);   // round-to-nearest-even
    return (unsigned short)(r >> 16);
}
static __device__ __forceinline__ float bf2f(unsigned short h) {
    return __uint_as_float(((unsigned int)h) << 16);
}

// ---------------------------------------------------------------------------
// Utility kernels
// ---------------------------------------------------------------------------
__global__ void zero_b128(uint4* p, long n) {
    long i = (long)blockIdx.x * blockDim.x + threadIdx.x;
    if (i < n) p[i] = make_uint4(0u, 0u, 0u, 0u);
}

// Scatter sparse voxel features into dense NDHWC bf16 grid + set mask
__global__ void scatter_kernel(const float* __restrict__ feats,
                               const int* __restrict__ coors,
                               unsigned short* __restrict__ dense,
                               unsigned char* __restrict__ mask, int n) {
    int i = blockIdx.x * blockDim.x + threadIdx.x;
    if (i >= n) return;
    int b = coors[i * 4 + 0], z = coors[i * 4 + 1];
    int y = coors[i * 4 + 2], x = coors[i * 4 + 3];
    size_t base = ((((size_t)b * 41 + z) * 200 + y) * 176 + x);
    unsigned short* dst = dense + base * 16;
    const float* src = feats + (size_t)i * 16;
#pragma unroll
    for (int c = 0; c < 16; ++c) dst[c] = f2bf(src[c]);
    mask[base] = 1;
}

// Max-pool (OR) the occupancy mask for a stride layer
__global__ void maskpool_kernel(const unsigned char* __restrict__ mi,
                                unsigned char* __restrict__ mo,
                                int Di, int Hi, int Wi, int Do, int Ho, int Wo,
                                int kd, int kh, int kw, int sd, int sh, int sw,
                                int pd, int ph, int pw) {
    int idx = blockIdx.x * blockDim.x + threadIdx.x;
    int total = 2 * Do * Ho * Wo;
    if (idx >= total) return;
    int x = idx % Wo;
    int y = (idx / Wo) % Ho;
    int z = (idx / (Wo * Ho)) % Do;
    int b = idx / (Wo * Ho * Do);
    unsigned char v = 0;
    for (int dz = 0; dz < kd; ++dz) {
        int zi = z * sd + dz - pd; if (zi < 0 || zi >= Di) continue;
        for (int dy = 0; dy < kh; ++dy) {
            int yi = y * sh + dy - ph; if (yi < 0 || yi >= Hi) continue;
            for (int dx = 0; dx < kw; ++dx) {
                int xi = x * sw + dx - pw; if (xi < 0 || xi >= Wi) continue;
                v |= mi[(((size_t)b * Di + zi) * Hi + yi) * Wi + xi];
            }
        }
    }
    mo[idx] = v ? 1 : 0;
}

// Pre-swizzle fp32 DHWIO weights into bf16 WMMA-B fragments:
// dst[((chunk*NT + ntile)*32 + lane)*16 + j],
// lane: n = ntile*16 + (lane&15), h = lane>>4 ; element j -> K = 16*h + j
__global__ void prep_weights(const float* __restrict__ w,
                             unsigned short* __restrict__ dst,
                             int kd, int kh, int kw, int cin, int cout, int nchunk) {
    int idx = blockIdx.x * blockDim.x + threadIdx.x;
    int nt = cout / 16;
    int total = nchunk * nt * 512;
    if (idx >= total) return;
    int j = idx & 15;
    int lane = (idx >> 4) & 31;
    int t = idx >> 9;             // chunk*nt + ntile
    int ntile = t % nt;
    int c = t / nt;
    int h = lane >> 4;
    int n = ntile * 16 + (lane & 15);
    int k = 16 * h + j;
    int kk = c * 32 + k;
    int ktot = kd * kh * kw * cin;
    float val = 0.f;
    if (kk < ktot) {
        int tap = kk / cin, ci = kk % cin;
        int dz = tap / (kh * kw), dy = (tap / kw) % kh, dx = tap % kw;
        val = w[(((size_t)((dz * kh + dy) * kw + dx) * cin + ci) * cout) + n];
    }
    dst[idx] = f2bf(val);
}

// ---------------------------------------------------------------------------
// Implicit-GEMM conv3d + BN + ReLU + mask, bf16 WMMA (16x16x32)
// One block: 16*MT output x-positions at (b,z,y); wave nt owns couts
// [nt*16, nt*16+16). Each B fragment is reused for MT WMMAs (MT accumulators).
// K = kd*kh*kw*cin, tap-major/channel-minor, chunks of 32.
// ---------------------------------------------------------------------------
template <int CIN, int COUT, int MT, int KD_, int KH_, int KW_,
          int SD_, int SH_, int SW_, int PD_, int PH_, int PW_>
__global__ void __launch_bounds__(32 * (COUT / 16))
conv3d_bn_relu_mask(const unsigned short* __restrict__ in,
                    const unsigned short* __restrict__ wswz,
                    const float* __restrict__ gamma, const float* __restrict__ beta,
                    const float* __restrict__ mean,  const float* __restrict__ var,
                    const unsigned char* __restrict__ mask,
                    unsigned short* __restrict__ out,
                    int Di, int Hi, int Wi, int Do, int Ho, int Wo) {
    constexpr int NT     = COUT / 16;
    constexpr int KTOT   = KD_ * KH_ * KW_ * CIN;
    constexpr int NCHUNK = (KTOT + 31) / 32;
    constexpr int XSPAN  = (16 * MT - 1) * SW_ + KW_;
    constexpr int PATCH  = KD_ * KH_ * XSPAN * CIN;
    constexpr int TSTEP  = 16 * SW_ * CIN;      // LDS element step between M tiles
    __shared__ unsigned short lds_patch[PATCH];

    const int x0 = blockIdx.x * (16 * MT);
    const int y  = blockIdx.y;
    const int z  = blockIdx.z % Do;
    const int b  = blockIdx.z / Do;

    const int tid  = threadIdx.x;
    const int lane = tid & 31;
    const int nt   = tid >> 5;
    const int h    = lane >> 4;
    const int m16  = lane & 15;

    // ---- cooperative patch load: global b128 -> LDS b128, zero halo ----
    for (int seg = 0; seg < KD_ * KH_; ++seg) {
        const int dz = seg / KH_, dy = seg % KH_;
        const int zi = z * SD_ + dz - PD_;
        const int yi = y * SH_ + dy - PH_;
        const bool rowok = ((unsigned)zi < (unsigned)Di) && ((unsigned)yi < (unsigned)Hi);
        const unsigned short* srow =
            in + ((((size_t)b * Di + (rowok ? zi : 0)) * Hi + (rowok ? yi : 0)) * (size_t)Wi) * CIN;
        for (int e = tid; e < (XSPAN * CIN) / 8; e += 32 * NT) {
            const int xoff = (e * 8) / CIN;
            const int ci   = (e * 8) % CIN;
            const int xi   = x0 * SW_ + xoff - PW_;
            v8us v = {0, 0, 0, 0, 0, 0, 0, 0};
            if (rowok && (unsigned)xi < (unsigned)Wi)
                v = *(const v8us*)(srow + (size_t)xi * CIN + ci);
            *(v8us*)(&lds_patch[(seg * XSPAN + xoff) * CIN + ci]) = v;
        }
    }
    __syncthreads();

    // ---- K loop: B once per chunk (2x global_load_b128, L2-resident),
    //      A per M-tile (2x ds_load_b128), MT WMMAs per chunk
    v8f acc[MT];
#pragma unroll
    for (int mt = 0; mt < MT; ++mt)
#pragma unroll
        for (int r = 0; r < 8; ++r) acc[mt][r] = 0.f;

    const unsigned short* wb = wswz + (size_t)(nt * 512 + lane * 16);
    for (int c = 0; c < NCHUNK; ++c) {
        const v8us* bp = (const v8us*)(wb + (size_t)c * (NT * 512));
        v8us b0 = bp[0];
        v8us b1 = bp[1];
        v16us bu = __builtin_shufflevector(b0, b1, 0, 1, 2, 3, 4, 5, 6, 7,
                                           8, 9, 10, 11, 12, 13, 14, 15);
        v16bf bv = __builtin_bit_cast(v16bf, bu);

        // A lane layout (16-bit 16x32): element j -> K = 16*h + j (matches B)
        const int kk0 = c * 32 + 16 * h;       // elements j=0..7
        const int kk1 = kk0 + 8;               // elements j=8..15
        int off0 = -1, off1 = -1;
        if (kk0 < KTOT) {
            int tap = kk0 / CIN, ci = kk0 % CIN;
            int dz = tap / (KH_ * KW_), dy = (tap / KW_) % KH_, dx = tap % KW_;
            off0 = ((dz * KH_ + dy) * XSPAN + m16 * SW_ + dx) * CIN + ci;
        }
        if (kk1 < KTOT) {
            int tap = kk1 / CIN, ci = kk1 % CIN;
            int dz = tap / (KH_ * KW_), dy = (tap / KW_) % KH_, dx = tap % KW_;
            off1 = ((dz * KH_ + dy) * XSPAN + m16 * SW_ + dx) * CIN + ci;
        }
#pragma unroll
        for (int mt = 0; mt < MT; ++mt) {
            v8us a0 = {0, 0, 0, 0, 0, 0, 0, 0};
            v8us a1 = {0, 0, 0, 0, 0, 0, 0, 0};
            if (off0 >= 0) a0 = *(const v8us*)(&lds_patch[off0 + mt * TSTEP]);
            if (off1 >= 0) a1 = *(const v8us*)(&lds_patch[off1 + mt * TSTEP]);
            v16us au = __builtin_shufflevector(a0, a1, 0, 1, 2, 3, 4, 5, 6, 7,
                                               8, 9, 10, 11, 12, 13, 14, 15);
            v16bf av = __builtin_bit_cast(v16bf, au);
            acc[mt] = __builtin_amdgcn_wmma_f32_16x16x32_bf16(
                false, av, false, bv, (short)0, acc[mt], false, false);
        }
    }

    // ---- epilogue: BN + ReLU + mask, bf16 store (lane-contiguous channels)
    const int n = nt * 16 + m16;
    const float scv = gamma[n] * rsqrtf(var[n] + 1e-3f);
    const float shv = beta[n] - mean[n] * scv;
    const size_t orow = (((size_t)b * Do + z) * Ho + y) * (size_t)Wo;
#pragma unroll
    for (int mt = 0; mt < MT; ++mt) {
#pragma unroll
        for (int r = 0; r < 8; ++r) {
            const int m = r + 8 * h;   // D layout: vgpr r holds M=r (+8 for lanes 16-31)
            const int x = x0 + mt * 16 + m;
            if (x < Wo) {
                float v = acc[mt][r] * scv + shv;
                v = fmaxf(v, 0.f);
                v *= (float)mask[orow + x];
                out[(orow + x) * COUT + n] = f2bf(v);
            }
        }
    }
}

// NDHWC bf16 (2, D, 25, 22, 64) -> NCDHW fp32 flat
__global__ void pack_out(const unsigned short* __restrict__ feat,
                         float* __restrict__ outp, int D, int total) {
    int idx = blockIdx.x * blockDim.x + threadIdx.x;
    if (idx >= total) return;
    int x = idx % 22;
    int y = (idx / 22) % 25;
    int z = (idx / (22 * 25)) % D;
    int c = (idx / (22 * 25 * D)) % 64;
    int b = idx / (22 * 25 * D * 64);
    outp[idx] = bf2f(feat[((((size_t)b * D + z) * 25 + y) * 22 + x) * 64 + c]);
}

// ---------------------------------------------------------------------------
// Host orchestration
// ---------------------------------------------------------------------------
struct LC { int cin, cout, kd, kh, kw, sd, sh, sw, pd, ph, pw; };
static const LC LCFG[14] = {
    {16, 16, 3, 3, 3, 1, 1, 1, 1, 1, 1}, {16, 16, 3, 3, 3, 1, 1, 1, 1, 1, 1},
    {16, 32, 3, 3, 3, 2, 2, 2, 1, 1, 1}, {32, 32, 3, 3, 3, 1, 1, 1, 1, 1, 1},
    {32, 32, 3, 3, 3, 1, 1, 1, 1, 1, 1}, {32, 64, 3, 3, 3, 2, 2, 2, 1, 1, 1},
    {64, 64, 3, 3, 3, 1, 1, 1, 1, 1, 1}, {64, 64, 3, 3, 3, 1, 1, 1, 1, 1, 1},
    {64, 64, 3, 3, 3, 1, 1, 1, 1, 1, 1}, {64, 64, 3, 3, 3, 2, 2, 2, 0, 1, 1},
    {64, 64, 3, 3, 3, 1, 1, 1, 1, 1, 1}, {64, 64, 3, 3, 3, 1, 1, 1, 1, 1, 1},
    {64, 64, 3, 3, 3, 1, 1, 1, 1, 1, 1}, {64, 64, 3, 1, 1, 2, 1, 1, 0, 0, 0},
};

extern "C" void kernel_launch(void* const* d_in, const int* in_sizes, int n_in,
                              void* d_out, int out_size, void* d_ws, size_t ws_size,
                              hipStream_t stream) {
    const float* vf    = (const float*)d_in[0];
    const int*   coors = (const int*)d_in[1];
    const float* Wt[14]; const float* Gm[14]; const float* Bt[14];
    const float* Mn[14]; const float* Vr[14];
    for (int i = 0; i < 14; ++i) {
        Wt[i] = (const float*)d_in[4 + i];
        Gm[i] = (const float*)d_in[18 + i];
        Bt[i] = (const float*)d_in[32 + i];
        Mn[i] = (const float*)d_in[46 + i];
        Vr[i] = (const float*)d_in[60 + i];
    }

    // ws carve-up
    char* ws = (char*)d_ws;
    size_t off = 0;
    auto carve = [&](size_t bytes) -> void* {
        void* p = ws + off;
        off += (bytes + 255) & ~(size_t)255;
        return p;
    };
    const size_t FEAT_ELEMS = (size_t)2 * 41 * 200 * 176 * 16;   // 46,182,400
    unsigned short* bufA = (unsigned short*)carve(FEAT_ELEMS * 2);
    unsigned short* bufB = (unsigned short*)carve(FEAT_ELEMS * 2);
    unsigned char*  m0   = (unsigned char*)carve((size_t)2 * 41 * 200 * 176);
    unsigned char*  m1   = (unsigned char*)carve((size_t)2 * 41 * 200 * 176);
    unsigned short* wz[14];
    int nchunk[14];
    for (int i = 0; i < 14; ++i) {
        const LC& L = LCFG[i];
        nchunk[i] = (L.kd * L.kh * L.kw * L.cin + 31) / 32;
        size_t elems = (size_t)nchunk[i] * (L.cout / 16) * 512;
        wz[i] = (unsigned short*)carve(elems * 2);
    }

    // zero the initial dense grid + mask
    {
        long n = (long)(FEAT_ELEMS * 2 / 16);
        zero_b128<<<dim3((unsigned)((n + 255) / 256)), 256, 0, stream>>>((uint4*)bufA, n);
        long nm = (long)((size_t)2 * 41 * 200 * 176 / 16);
        zero_b128<<<dim3((unsigned)((nm + 255) / 256)), 256, 0, stream>>>((uint4*)m0, nm);
    }

    // scatter voxels
    int nvox = in_sizes[1] / 4;
    scatter_kernel<<<dim3((nvox + 255) / 256), 256, 0, stream>>>(vf, coors, bufA, m0, nvox);

    // weight swizzle
    for (int i = 0; i < 14; ++i) {
        const LC& L = LCFG[i];
        int tot = nchunk[i] * (L.cout / 16) * 512;
        prep_weights<<<dim3((tot + 255) / 256), 256, 0, stream>>>(
            Wt[i], wz[i], L.kd, L.kh, L.kw, L.cin, L.cout, nchunk[i]);
    }

#define CONV(i, CI, CO, MT, KD, KH, KW, SD, SH, SW, PD, PH, PW, src, dst, msk, Di, Hi, Wi, Dd, Hh, Ww) \
    conv3d_bn_relu_mask<CI, CO, MT, KD, KH, KW, SD, SH, SW, PD, PH, PW>                                \
        <<<dim3(((Ww) + 16 * (MT)-1) / (16 * (MT)), (Hh), (Dd)*2), 32 * ((CO) / 16), 0, stream>>>(     \
            src, wz[i], Gm[i], Bt[i], Mn[i], Vr[i], msk, dst, Di, Hi, Wi, Dd, Hh, Ww)

    // stage 1: 41 x 200 x 176, C=16  (MT=2: B reuse across 32-wide x tiles)
    CONV(0, 16, 16, 2, 3, 3, 3, 1, 1, 1, 1, 1, 1, bufA, bufB, m0, 41, 200, 176, 41, 200, 176);
    CONV(1, 16, 16, 2, 3, 3, 3, 1, 1, 1, 1, 1, 1, bufB, bufA, m0, 41, 200, 176, 41, 200, 176);

    // downsample -> 21 x 100 x 88, C=32
    maskpool_kernel<<<dim3((2 * 21 * 100 * 88 + 255) / 256), 256, 0, stream>>>(
        m0, m1, 41, 200, 176, 21, 100, 88, 3, 3, 3, 2, 2, 2, 1, 1, 1);
    CONV(2, 16, 32, 2, 3, 3, 3, 2, 2, 2, 1, 1, 1, bufA, bufB, m1, 41, 200, 176, 21, 100, 88);
    CONV(3, 32, 32, 2, 3, 3, 3, 1, 1, 1, 1, 1, 1, bufB, bufA, m1, 21, 100, 88, 21, 100, 88);
    CONV(4, 32, 32, 2, 3, 3, 3, 1, 1, 1, 1, 1, 1, bufA, bufB, m1, 21, 100, 88, 21, 100, 88);

    // downsample -> 11 x 50 x 44, C=64  (MT=1: keep block-level parallelism)
    maskpool_kernel<<<dim3((2 * 11 * 50 * 44 + 255) / 256), 256, 0, stream>>>(
        m1, m0, 21, 100, 88, 11, 50, 44, 3, 3, 3, 2, 2, 2, 1, 1, 1);
    CONV(5, 32, 64, 1, 3, 3, 3, 2, 2, 2, 1, 1, 1, bufB, bufA, m0, 21, 100, 88, 11, 50, 44);
    CONV(6, 64, 64, 1, 3, 3, 3, 1, 1, 1, 1, 1, 1, bufA, bufB, m0, 11, 50, 44, 11, 50, 44);
    CONV(7, 64, 64, 1, 3, 3, 3, 1, 1, 1, 1, 1, 1, bufB, bufA, m0, 11, 50, 44, 11, 50, 44);
    CONV(8, 64, 64, 1, 3, 3, 3, 1, 1, 1, 1, 1, 1, bufA, bufB, m0, 11, 50, 44, 11, 50, 44);

    // downsample -> 5 x 25 x 22, C=64  (pad (0,1,1))
    maskpool_kernel<<<dim3((2 * 5 * 25 * 22 + 255) / 256), 256, 0, stream>>>(
        m0, m1, 11, 50, 44, 5, 25, 22, 3, 3, 3, 2, 2, 2, 0, 1, 1);
    CONV(9, 64, 64, 1, 3, 3, 3, 2, 2, 2, 0, 1, 1, bufB, bufA, m1, 11, 50, 44, 5, 25, 22);
    CONV(10, 64, 64, 1, 3, 3, 3, 1, 1, 1, 1, 1, 1, bufA, bufB, m1, 5, 25, 22, 5, 25, 22);
    CONV(11, 64, 64, 1, 3, 3, 3, 1, 1, 1, 1, 1, 1, bufB, bufA, m1, 5, 25, 22, 5, 25, 22);
    CONV(12, 64, 64, 1, 3, 3, 3, 1, 1, 1, 1, 1, 1, bufA, bufB, m1, 5, 25, 22, 5, 25, 22);

    // conv_4 output (layer 12 activation): (2,64,5,25,22) fp32 at offset 140800
    pack_out<<<dim3((352000 + 255) / 256), 256, 0, stream>>>(
        bufB, (float*)d_out + 140800, 5, 352000);

    // final layer: k=(3,1,1), s=(2,1,1), p=0 -> 2 x 25 x 22
    maskpool_kernel<<<dim3((2 * 2 * 25 * 22 + 255) / 256), 256, 0, stream>>>(
        m1, m0, 5, 25, 22, 2, 25, 22, 3, 1, 1, 2, 1, 1, 0, 0, 0);
    CONV(13, 64, 64, 1, 3, 1, 1, 2, 1, 1, 0, 0, 0, bufB, bufA, m0, 5, 25, 22, 2, 25, 22);

    // ret output: (2,128,25,22) fp32 == NCDHW flat of (2,64,2,25,22)
    pack_out<<<dim3((140800 + 255) / 256), 256, 0, stream>>>(
        bufA, (float*)d_out, 2, 140800);
#undef CONV
}
